// PAM_Module_35588099015598
// MI455X (gfx1250) — matrix operand order repaired and checked by reading, not compiled, not run on hardware
//
#include <hip/hip_runtime.h>
#include <math.h>

// Problem constants (match reference)
#define BN 8
#define CC 512
#define NN 4096   // HW*HW
#define MM 64     // C/SCALE
#define EPSF 1e-6f

typedef __attribute__((ext_vector_type(16))) __bf16 v16bf;
typedef __attribute__((ext_vector_type(8)))  __bf16 v8bf;
typedef __attribute__((ext_vector_type(8)))  float  v8f;

// ---------------------------------------------------------------------------
// WMMA helper: D = A(16x32 bf16) x B(32x16 bf16) + C(16x16 f32)
// ---------------------------------------------------------------------------
__device__ __forceinline__ v8f wmma_bf16(v16bf a, v16bf b, v8f c) {
  return __builtin_amdgcn_wmma_f32_16x16x32_bf16(
      /*neg_a=*/false, a, /*neg_b=*/false, b,
      /*c_mod=*/(short)0, c, /*reuse_a=*/false, /*reuse_b=*/false);
}

__device__ __forceinline__ v16bf cat16(v8bf lo, v8bf hi) {
  return __builtin_shufflevector(lo, hi, 0, 1, 2, 3, 4, 5, 6, 7,
                                 8, 9, 10, 11, 12, 13, 14, 15);
}

// A-matrix (16x32 bf16) fragment for lane (hi = lane>=16, row = m0 + lane%16):
//   elements 0..7  = K[hi*8     .. hi*8+7]   (contiguous)
//   elements 8..15 = K[16+hi*8 .. 16+hi*8+7] (contiguous)
// row must be a bf16* to the start of the K-window (16B aligned chunks).
__device__ __forceinline__ v16bf load_a_frag(const __bf16* row, int hi) {
  const __bf16* p = row + hi * 8;
  v8bf lo = *(const v8bf*)p;
  v8bf hi8 = *(const v8bf*)(p + 16);
  return cat16(lo, hi8);
}

// B-matrix (32x16 bf16) fragment for lane (col = lane%16):
//   elements 0..7  = K[hi*8     .. hi*8+7]   (contiguous)
//   elements 8..15 = K[16+hi*8 .. 16+hi*8+7] (contiguous)
__device__ __forceinline__ v16bf load_b_frag(const __bf16* col, int hi) {
  const __bf16* p = col + hi * 8;
  v8bf lo = *(const v8bf*)p;
  v8bf hi8 = *(const v8bf*)(p + 16);
  return cat16(lo, hi8);
}

// Fast softplus via hardware transcendentals (v_exp_f32 / v_log_f32):
// softplus(x) = max(x,0) + log(1 + exp(-|x|))
__device__ __forceinline__ float softplus_fast(float x) {
  float t = __expf(-fabsf(x));
  return fmaxf(x, 0.0f) + __logf(1.0f + t);
}

// ---------------------------------------------------------------------------
// Kernel 0: convert weights fp32 -> bf16
// ---------------------------------------------------------------------------
__global__ void k_convert_w(const float* __restrict__ wq, const float* __restrict__ wk,
                            const float* __restrict__ wv,
                            __bf16* __restrict__ wqb, __bf16* __restrict__ wkb,
                            __bf16* __restrict__ wvb) {
  int i = blockIdx.x * blockDim.x + threadIdx.x;
  if (i < MM * CC) { wqb[i] = (__bf16)wq[i]; wkb[i] = (__bf16)wk[i]; }
  if (i < CC * CC) { wvb[i] = (__bf16)wv[i]; }
}

// ---------------------------------------------------------------------------
// Kernel 1: per (b,c) row: gate = max_n + mean_n, and x -> bf16 copy
// ---------------------------------------------------------------------------
__global__ void k_gate_convert(const float* __restrict__ x, __bf16* __restrict__ xb,
                               float* __restrict__ gate) {
  const int row = blockIdx.x;                 // b*CC + c
  const float* xr = x + (size_t)row * NN;
  __bf16* xbr = xb + (size_t)row * NN;
  float mx = -1e30f, sm = 0.0f;
  for (int i = threadIdx.x; i < NN; i += blockDim.x) {
    float v = xr[i];
    xbr[i] = (__bf16)v;
    mx = fmaxf(mx, v);
    sm += v;
  }
  __shared__ float smax[256], ssum[256];
  smax[threadIdx.x] = mx; ssum[threadIdx.x] = sm;
  __syncthreads();
  for (int s = blockDim.x >> 1; s > 0; s >>= 1) {
    if ((int)threadIdx.x < s) {
      smax[threadIdx.x] = fmaxf(smax[threadIdx.x], smax[threadIdx.x + s]);
      ssum[threadIdx.x] += ssum[threadIdx.x + s];
    }
    __syncthreads();
  }
  if (threadIdx.x == 0) gate[row] = smax[0] + ssum[0] * (1.0f / NN);
}

// ---------------------------------------------------------------------------
// Kernel 2: LDS-tiled transpose xb [C,N] -> xbT [N,C] (bf16), 64x64 tiles.
// Coalesced on both sides; LDS row padded to 72 to break bank conflicts.
// ---------------------------------------------------------------------------
__global__ void __launch_bounds__(256)
k_transpose(const __bf16* __restrict__ xb, __bf16* __restrict__ xbT) {
  __shared__ __bf16 tile[64][72];
  const int b  = blockIdx.z;
  const int cb = blockIdx.y * 64;
  const int nb = blockIdx.x * 64;
  const __bf16* src = xb + (size_t)b * CC * NN;
  __bf16* dst = xbT + (size_t)b * NN * CC;

  // Load: thread t covers row r = t/4, 16 contiguous n at (t%4)*16.
  {
    int r = threadIdx.x >> 2;
    int cg = (threadIdx.x & 3) * 16;
    const __bf16* p = src + (size_t)(cb + r) * NN + nb + cg;
    *(v8bf*)&tile[r][cg]     = *(const v8bf*)p;
    *(v8bf*)&tile[r][cg + 8] = *(const v8bf*)(p + 8);
  }
  __syncthreads();
  // Store: thread t covers output row n = t/4, 16 contiguous c at (t%4)*16.
  {
    int n = threadIdx.x >> 2;
    int cg = (threadIdx.x & 3) * 16;
    v8bf v0, v1;
#pragma unroll
    for (int i = 0; i < 8; ++i) v0[i] = tile[cg + i][n];
#pragma unroll
    for (int i = 0; i < 8; ++i) v1[i] = tile[cg + 8 + i][n];
    __bf16* p = dst + (size_t)(nb + n) * CC + cb + cg;
    *(v8bf*)p = v0;
    *(v8bf*)(p + 8) = v1;
  }
}

// ---------------------------------------------------------------------------
// Kernel 3: Q/K projection GEMM. One wave -> 16(M) x 64(N) tile for Q AND K.
// A = weights [M,C] (contiguous), B = xbT [N,C] (contiguous), K-steps of 32.
// Outputs: QfT [B,N,M] (b128 stores per lane), Kf [B,M,N].
// ---------------------------------------------------------------------------
__global__ void __launch_bounds__(32)
k_qk(const __bf16* __restrict__ xbT, const __bf16* __restrict__ wqb,
     const __bf16* __restrict__ wkb, const float* __restrict__ bq,
     const float* __restrict__ bk, __bf16* __restrict__ QfT,
     __bf16* __restrict__ Kf) {
  const int lane = threadIdx.x, hi = lane >> 4, lm = lane & 15;
  const int n0 = blockIdx.x * 64;
  const int m0 = blockIdx.y * 16;
  const int b  = blockIdx.z;
  const __bf16* xT = xbT + (size_t)b * NN * CC;

  v8f accq[4] = {}; v8f acck[4] = {};
  for (int c0 = 0; c0 < CC; c0 += 32) {
    const __bf16* aqp = wqb + (size_t)(m0 + lm) * CC + c0;
    const __bf16* akp = wkb + (size_t)(m0 + lm) * CC + c0;
    __builtin_prefetch(aqp + 32, 0, 0);
    v16bf aq = load_a_frag(aqp, hi);
    v16bf ak = load_a_frag(akp, hi);
#pragma unroll
    for (int j = 0; j < 4; ++j) {
      const __bf16* bp = xT + (size_t)(n0 + j * 16 + lm) * CC + c0;
      __builtin_prefetch(bp + 32, 0, 0);
      v16bf bf = load_b_frag(bp, hi);
      accq[j] = wmma_bf16(aq, bf, accq[j]);
      acck[j] = wmma_bf16(ak, bf, acck[j]);
    }
  }
#pragma unroll
  for (int j = 0; j < 4; ++j) {
    const int n = n0 + j * 16 + lm;
    v8bf qv;
#pragma unroll
    for (int r = 0; r < 8; ++r) {
      int m = m0 + hi * 8 + r;
      qv[r] = (__bf16)softplus_fast(accq[j][r] + bq[m]);
      Kf[((size_t)b * MM + m) * NN + n] =
          (__bf16)softplus_fast(acck[j][r] + bk[m]);
    }
    *(v8bf*)(QfT + ((size_t)b * NN + n) * MM + m0 + hi * 8) = qv;
  }
}

// ---------------------------------------------------------------------------
// Kernel 4: ksum[b,m] = sum_n Kf[b,m,n]
// ---------------------------------------------------------------------------
__global__ void k_ksum(const __bf16* __restrict__ Kf, float* __restrict__ ksum) {
  const int row = blockIdx.x;                 // b*MM + m
  const __bf16* kr = Kf + (size_t)row * NN;
  float s = 0.0f;
  for (int i = threadIdx.x; i < NN; i += blockDim.x) s += (float)kr[i];
  __shared__ float sh[256];
  sh[threadIdx.x] = s; __syncthreads();
  for (int st = blockDim.x >> 1; st > 0; st >>= 1) {
    if ((int)threadIdx.x < st) sh[threadIdx.x] += sh[threadIdx.x + st];
    __syncthreads();
  }
  if (threadIdx.x == 0) ksum[row] = sh[0];
}

// ---------------------------------------------------------------------------
// Kernel 5: KX[b,m,c'] = sum_n Kf[b,m,n] * x[b,c',n]  (contraction over N=4096)
// A = Kf rows (contiguous), B = xb rows (contiguous). 16(M) x 64(C') per wave.
// ---------------------------------------------------------------------------
__global__ void __launch_bounds__(32)
k_kx(const __bf16* __restrict__ Kf, const __bf16* __restrict__ xb,
     __bf16* __restrict__ KX) {
  const int lane = threadIdx.x, hi = lane >> 4, lm = lane & 15;
  const int c0 = blockIdx.x * 64;
  const int m0 = blockIdx.y * 16;
  const int b  = blockIdx.z;
  const __bf16* Kb  = Kf + (size_t)b * MM * NN;
  const __bf16* xbb = xb + (size_t)b * CC * NN;

  v8f acc[4] = {};
  for (int n0 = 0; n0 < NN; n0 += 32) {
    const __bf16* ap = Kb + (size_t)(m0 + lm) * NN + n0;
    __builtin_prefetch(ap + 32, 0, 0);
    v16bf a = load_a_frag(ap, hi);
#pragma unroll
    for (int j = 0; j < 4; ++j) {
      const __bf16* bp = xbb + (size_t)(c0 + j * 16 + lm) * NN + n0;
      __builtin_prefetch(bp + 32, 0, 0);
      v16bf bf = load_b_frag(bp, hi);
      acc[j] = wmma_bf16(a, bf, acc[j]);
    }
  }
#pragma unroll
  for (int j = 0; j < 4; ++j)
#pragma unroll
    for (int r = 0; r < 8; ++r) {
      int m = m0 + hi * 8 + r, c = c0 + j * 16 + lm;
      KX[((size_t)b * MM + m) * CC + c] = (__bf16)acc[j][r];
    }
}

// ---------------------------------------------------------------------------
// Kernel 6: KV[b,m,c] = gate[b,c]*(sum_{c'} KX[b,m,c']*wv[c,c'] + bv[c]*ksum[b,m])
// Stored transposed KVT[b,c,m] -- per-lane contiguous over m => b128 stores.
// ---------------------------------------------------------------------------
__global__ void __launch_bounds__(32)
k_kv(const __bf16* __restrict__ KX, const __bf16* __restrict__ wvb,
     const float* __restrict__ bv, const float* __restrict__ gate,
     const float* __restrict__ ksum, __bf16* __restrict__ KVT) {
  const int lane = threadIdx.x, hi = lane >> 4, lm = lane & 15;
  const int c0 = blockIdx.x * 64;
  const int m0 = blockIdx.y * 16;
  const int b  = blockIdx.z;
  const __bf16* A = KX + (size_t)b * MM * CC;

  v8f acc[4] = {};
  for (int k0 = 0; k0 < CC; k0 += 32) {
    v16bf a = load_a_frag(A + (size_t)(m0 + lm) * CC + k0, hi);
#pragma unroll
    for (int j = 0; j < 4; ++j) {
      v16bf bf = load_b_frag(wvb + (size_t)(c0 + j * 16 + lm) * CC + k0, hi);
      acc[j] = wmma_bf16(a, bf, acc[j]);
    }
  }
#pragma unroll
  for (int j = 0; j < 4; ++j) {
    const int c = c0 + j * 16 + lm;
    const float g = gate[b * CC + c];
    const float bvc = bv[c];
    v8bf kv;
#pragma unroll
    for (int r = 0; r < 8; ++r) {
      int m = m0 + hi * 8 + r;
      kv[r] = (__bf16)(g * (acc[j][r] + bvc * ksum[b * MM + m]));
    }
    *(v8bf*)(KVT + ((size_t)b * CC + c) * MM + m0 + hi * 8) = kv;
  }
}

// ---------------------------------------------------------------------------
// Kernel 7: norm[b,n] = 1 / sum_m QfT[b,n,m] * (ksum[b,m] + EPS)
// Per-thread contiguous vector reads of QfT.
// ---------------------------------------------------------------------------
__global__ void k_norm(const __bf16* __restrict__ QfT, const float* __restrict__ ksum,
                       float* __restrict__ norm) {
  const int idx = blockIdx.x * blockDim.x + threadIdx.x;  // b*NN + n
  const int b = idx / NN;
  const __bf16* q = QfT + (size_t)idx * MM;
  const float* ks = ksum + b * MM;
  float d = 0.0f;
#pragma unroll
  for (int m8 = 0; m8 < MM; m8 += 8) {
    v8bf qv = *(const v8bf*)(q + m8);
#pragma unroll
    for (int r = 0; r < 8; ++r)
      d += (float)qv[r] * (ks[m8 + r] + EPSF);
  }
  norm[idx] = 1.0f / d;
}

// ---------------------------------------------------------------------------
// Kernel 8: out[b,c,n] = x[b,c,n] + gamma*norm[b,n]*sum_m KVT[b,c,m]*QfT[b,n,m]
// A = KVT rows (contiguous), B = QfT rows (contiguous). 16(C) x 64(N) per wave.
// ---------------------------------------------------------------------------
__global__ void __launch_bounds__(32)
k_out(const __bf16* __restrict__ KVT, const __bf16* __restrict__ QfT,
      const float* __restrict__ norm, const float* __restrict__ x,
      const float* __restrict__ gamma, float* __restrict__ out) {
  const int lane = threadIdx.x, hi = lane >> 4, lm = lane & 15;
  const int n0 = blockIdx.x * 64;
  const int c0 = blockIdx.y * 16;
  const int b  = blockIdx.z;
  const __bf16* KT = KVT + (size_t)b * CC * MM;
  const __bf16* QT = QfT + (size_t)b * NN * MM;
  const float g = gamma[0];

  v8f acc[4] = {};
#pragma unroll
  for (int k0 = 0; k0 < MM; k0 += 32) {
    v16bf a = load_a_frag(KT + (size_t)(c0 + lm) * MM + k0, hi);
#pragma unroll
    for (int j = 0; j < 4; ++j) {
      v16bf bf = load_b_frag(QT + (size_t)(n0 + j * 16 + lm) * MM + k0, hi);
      acc[j] = wmma_bf16(a, bf, acc[j]);
    }
  }
#pragma unroll
  for (int j = 0; j < 4; ++j)
#pragma unroll
    for (int r = 0; r < 8; ++r) {
      int c = c0 + hi * 8 + r, n = n0 + j * 16 + lm;
      size_t idx = ((size_t)b * CC + c) * NN + n;
      out[idx] = x[idx] + g * norm[b * NN + n] * acc[j][r];
    }
}

// ---------------------------------------------------------------------------
// Launcher
// ---------------------------------------------------------------------------
extern "C" void kernel_launch(void* const* d_in, const int* in_sizes, int n_in,
                              void* d_out, int out_size, void* d_ws, size_t ws_size,
                              hipStream_t stream) {
  (void)in_sizes; (void)n_in; (void)out_size; (void)ws_size;
  const float* x     = (const float*)d_in[0];
  const float* wq    = (const float*)d_in[1];
  const float* bq    = (const float*)d_in[2];
  const float* wk    = (const float*)d_in[3];
  const float* bk    = (const float*)d_in[4];
  const float* wv    = (const float*)d_in[5];
  const float* bv    = (const float*)d_in[6];
  const float* gamma = (const float*)d_in[7];
  float* out = (float*)d_out;

  // Workspace partition (~74 MB total)
  char* base = (char*)d_ws;
  size_t off = 0;
  auto alloc = [&](size_t bytes) -> char* {
    char* r = base + off;
    off = (off + bytes + 255) & ~(size_t)255;
    return r;
  };
  __bf16* xb   = (__bf16*)alloc((size_t)BN * CC * NN * 2);  // 32 MB
  __bf16* xbT  = (__bf16*)alloc((size_t)BN * NN * CC * 2);  // 32 MB
  __bf16* wqb  = (__bf16*)alloc((size_t)MM * CC * 2);
  __bf16* wkb  = (__bf16*)alloc((size_t)MM * CC * 2);
  __bf16* wvb  = (__bf16*)alloc((size_t)CC * CC * 2);
  float*  gate = (float*) alloc((size_t)BN * CC * 4);
  __bf16* QfT  = (__bf16*)alloc((size_t)BN * NN * MM * 2);  // 4 MB
  __bf16* Kf   = (__bf16*)alloc((size_t)BN * MM * NN * 2);  // 4 MB
  __bf16* KX   = (__bf16*)alloc((size_t)BN * MM * CC * 2);
  float*  ksum = (float*) alloc((size_t)BN * MM * 4);
  __bf16* KVT  = (__bf16*)alloc((size_t)BN * CC * MM * 2);
  float*  norm = (float*) alloc((size_t)BN * NN * 4);

  k_convert_w   <<<(CC * CC + 255) / 256, 256, 0, stream>>>(wq, wk, wv, wqb, wkb, wvb);
  k_gate_convert<<<BN * CC, 256, 0, stream>>>(x, xb, gate);
  k_transpose   <<<dim3(NN / 64, CC / 64, BN), 256, 0, stream>>>(xb, xbT);
  k_qk  <<<dim3(NN / 64, MM / 16, BN), 32, 0, stream>>>(xbT, wqb, wkb, bq, bk, QfT, Kf);
  k_ksum<<<BN * MM, 256, 0, stream>>>(Kf, ksum);
  k_kx  <<<dim3(CC / 64, MM / 16, BN), 32, 0, stream>>>(Kf, xb, KX);
  k_kv  <<<dim3(CC / 64, MM / 16, BN), 32, 0, stream>>>(KX, wvb, bv, gate, ksum, KVT);
  k_norm<<<(BN * NN) / 256, 256, 0, stream>>>(QfT, ksum, norm);
  k_out <<<dim3(NN / 64, CC / 16, BN), 32, 0, stream>>>(KVT, QfT, norm, x, gamma, out);
}
